// PerformerAttention_66743791780368
// MI455X (gfx1250) — matrix operand order repaired and checked by reading, compiled, not run on hardware
//
#include <hip/hip_runtime.h>

// ---------------- problem constants ----------------
#define B_  2
#define N_  1024
#define DIM_ 1024
#define H_  16
#define D_  64
#define M_  128
#define HD_ (H_*D_)      // 1024
#define BN_ (B_*N_)      // 2048
#define EPS_ 1e-6f
#define SCALE_ 0.35355339059327373f   // D^-0.25 = 64^-0.25
#define RATIO_ 0.08838834764831845f   // 1/sqrt(M) = 1/sqrt(128)

typedef __attribute__((ext_vector_type(16))) __bf16 bf16x16;
typedef __attribute__((ext_vector_type(8)))  float  floatx8;
typedef __attribute__((ext_vector_type(4)))  unsigned int uint32x4_;
typedef __attribute__((ext_vector_type(8)))  int  int32x8_;
typedef __attribute__((ext_vector_type(4)))  int  int32x4_;

union Frag { bf16x16 v; uint4 q[2]; };

#if __has_builtin(__builtin_amdgcn_tensor_load_to_lds)
#define HAVE_TDM 1
#else
#define HAVE_TDM 0
#endif

// GEMM tiling: block tile 128 rows x 128 cols, K-step 32, 8 waves.
// wave = (wr 0..3: 32-row strip) x (wc 0..1: 64-col half) -> 2x4 = 8 WMMA tiles/wave.
#define TM_ 128
#define TN_ 128
#define TK_ 32
#define A_BYTES_ (TM_ * TK_ * 2)            // 8 KB
#define B_BYTES_ (TN_ * TK_ * 2)            // 8 KB
#define STAGE_BYTES_ (A_BYTES_ + B_BYTES_)  // 16 KB
#define SMEM_BYTES_ (2 * STAGE_BYTES_)      // 32 KB (double buffered)

// ---------------- elementwise f32 -> bf16 (optional scale) ----------------
__global__ __launch_bounds__(256) void cvt_f32_bf16(const float* __restrict__ in,
                                                    __bf16* __restrict__ out,
                                                    int n, float scale) {
  int i = blockIdx.x * blockDim.x + threadIdx.x;
  if (i < n) out[i] = (__bf16)(in[i] * scale);
}

#if HAVE_TDM
// Build a 2-D Tensor-DMA descriptor (D#) and issue TENSOR_LOAD_TO_LDS.
// data_size = 2 bytes (bf16). tile/tensor dims in elements, stride0 in elements.
__device__ __forceinline__ void tdm_load_2d(unsigned lds_byte_off, const void* gaddr,
                                            unsigned tile_d0, unsigned tile_d1,
                                            unsigned long stride0) {
  unsigned long ga = (unsigned long)(uintptr_t)gaddr;
  uint32x4_ g0;
  g0[0] = 1u;                                               // count=1 (valid user D#)
  g0[1] = lds_byte_off;                                     // [63:32]  lds_addr
  g0[2] = (unsigned)ga;                                     // [95:64]  global_addr lo
  g0[3] = (unsigned)((ga >> 32) & 0x01FFFFFFu) | (2u << 30);// global_addr hi | type=2
  int32x8_ g1;
  g1[0] = (int)(1u << 16);                                  // data_size=1 -> 2B/elem
  g1[1] = (int)((tile_d0 & 0xFFFFu) << 16);                 // tensor_dim0 = tile_d0 (lo16)
  g1[2] = (int)(((tile_d0 >> 16) & 0xFFFFu) |
                ((tile_d1 & 0xFFFFu) << 16));               // t_dim0 hi | tensor_dim1 lo
  g1[3] = (int)(((tile_d1 >> 16) & 0xFFFFu) |
                ((tile_d0 & 0xFFFFu) << 16));               // t_dim1 hi | tile_dim0
  g1[4] = (int)(tile_d1 & 0xFFFFu);                         // tile_dim1 | tile_dim2=0
  g1[5] = (int)(unsigned)stride0;                           // tensor_dim0_stride lo32
  g1[6] = (int)((unsigned)(stride0 >> 32) & 0xFFFFu);       // stride0 hi16 | stride1 lo=0
  g1[7] = 0;
  int32x4_ z4 = {0, 0, 0, 0};
  int32x8_ z8 = {0, 0, 0, 0, 0, 0, 0, 0};
  __builtin_amdgcn_tensor_load_to_lds(g0, g1, z4, z4, z8, 0);
}
#endif

// One k-step of the block tile from a compile-time stage offset.
// Loads 2 A frags + 4 B frags (all as 16-B ds_load_b128 pairs), then 8 WMMAs.
template <int STAGE_OFF>
__device__ __forceinline__ void mm_stage(const char* smem, int wr, int wc, int lane,
                                         floatx8 (&acc)[2][4]) {
  const __bf16 (*As)[TK_] = (const __bf16(*)[TK_])(smem + STAGE_OFF);
  const __bf16 (*Bs)[TK_] = (const __bf16(*)[TK_])(smem + STAGE_OFF + A_BYTES_);

  // A fragment: 16-bit A 16x32 layout (lanes 0-15: K 0-7 & 16-23; lanes 16-31: K 8-15 & 24-31)
  // -> per lane exactly two contiguous 16-byte runs at K=khalf and K=16+khalf.
  const int khalf = (lane >> 4) * 8;
  Frag af[2];
#pragma unroll
  for (int s = 0; s < 2; ++s) {
    const int arow = wr * 32 + s * 16 + (lane & 15);
    af[s].q[0] = *reinterpret_cast<const uint4*>(&As[arow][khalf]);
    af[s].q[1] = *reinterpret_cast<const uint4*>(&As[arow][16 + khalf]);
  }

  // B fragments: 32x16, lane = column, lanes 0-15 hold K 0-7 & 16-23, lanes 16-31 hold K 8-15 & 24-31
  const int koff = khalf;
  Frag bf[4];
#pragma unroll
  for (int t = 0; t < 4; ++t) {
    const uint4* p = reinterpret_cast<const uint4*>(&Bs[wc * 64 + t * 16 + (lane & 15)][koff]);
    bf[t].q[0] = p[0];
    bf[t].q[1] = p[2];
  }

#pragma unroll
  for (int s = 0; s < 2; ++s)
#pragma unroll
    for (int t = 0; t < 4; ++t)
      acc[s][t] = __builtin_amdgcn_wmma_f32_16x16x32_bf16(false, af[s].v, false, bf[t].v,
                                                          (short)0, acc[s][t], false, false);
}

// ---------------- generic bf16 WMMA GEMM ----------------
// C[i,j] = sum_k A[i,k] * Bw[j,k]   (Bw row-major [Nc,K], i.e. "x @ W^T" form)
// TDM double-buffers LDS stages at compile-time-constant offsets; wave 0 drives the
// DMA, TENSORcnt gates the handoff, pipeline unrolled by 2 so no runtime buf index.
// blockIdx.z batching: A offset = (z/zdiv)*aZouter + (z%zdiv)*aZinner, C offset = z*cZ.
__global__ __launch_bounds__(256) void gemm_bf16_wmma(
    const __bf16* __restrict__ A, int lda,
    const __bf16* __restrict__ Bw, int ldb,
    float* __restrict__ C, int ldc, int K,
    int zdiv, long aZouter, long aZinner, long cZ,
    const float* __restrict__ bias)
{
  extern __shared__ char smem[];
  const int z = blockIdx.z;
  const __bf16* Ab = A + (long)(z / zdiv) * aZouter + (long)(z % zdiv) * aZinner;
  float* Cb = C + (long)z * cZ;
  const int row0 = blockIdx.y * TM_;
  const int col0 = blockIdx.x * TN_;

  const int tid  = threadIdx.x;
  const int lane = tid & 31;
  const int wave = tid >> 5;
  const int wr   = wave >> 1;   // 0..3: 32-row strip
  const int wc   = wave & 1;    // 0..1: 64-col half

  floatx8 acc[2][4] = {{{}, {}, {}, {}}, {{}, {}, {}, {}}};

  const int nk = K / TK_;
  const __bf16* Atile = Ab + (long)row0 * lda;
  const __bf16* Btile = Bw + (long)col0 * ldb;

#if HAVE_TDM
  const unsigned smem_base = (unsigned)(uintptr_t)(void*)smem;  // LDS byte offset
  if (wave == 0) {
    tdm_load_2d(smem_base,            Atile, TK_, TM_, (unsigned long)lda);
    tdm_load_2d(smem_base + A_BYTES_, Btile, TK_, TN_, (unsigned long)ldb);
  }
  for (int i = 0; i < nk; i += 2) {
    // ---- even stage: compute buf0, TDM prefetch into buf1 ----
    if (wave == 0) {
      if (i + 1 < nk) {
        const int kn = (i + 1) * TK_;
        tdm_load_2d(smem_base + STAGE_BYTES_,            Atile + kn, TK_, TM_, (unsigned long)lda);
        tdm_load_2d(smem_base + STAGE_BYTES_ + A_BYTES_, Btile + kn, TK_, TN_, (unsigned long)ldb);
        __builtin_amdgcn_s_wait_tensorcnt(2);
      } else {
        __builtin_amdgcn_s_wait_tensorcnt(0);
      }
    }
    __syncthreads();
    mm_stage<0>(smem, wr, wc, lane, acc);
    __syncthreads();
    if (i + 1 >= nk) break;

    // ---- odd stage: compute buf1, TDM prefetch into buf0 ----
    if (wave == 0) {
      if (i + 2 < nk) {
        const int kn = (i + 2) * TK_;
        tdm_load_2d(smem_base,            Atile + kn, TK_, TM_, (unsigned long)lda);
        tdm_load_2d(smem_base + A_BYTES_, Btile + kn, TK_, TN_, (unsigned long)ldb);
        __builtin_amdgcn_s_wait_tensorcnt(2);
      } else {
        __builtin_amdgcn_s_wait_tensorcnt(0);
      }
    }
    __syncthreads();
    mm_stage<STAGE_BYTES_>(smem, wr, wc, lane, acc);
    __syncthreads();
  }
#else
  for (int i = 0; i < nk; ++i) {
    // cooperative fallback: 256 threads stage A (128x32) + B (128x32), 32B each
    const int k0 = i * TK_;
    const int arow = tid >> 1, acol = (tid & 1) * 16;
    *reinterpret_cast<ulonglong4*>(smem + (arow * TK_ + acol) * 2) =
        *reinterpret_cast<const ulonglong4*>(Atile + (long)arow * lda + k0 + acol);
    *reinterpret_cast<ulonglong4*>(smem + A_BYTES_ + (arow * TK_ + acol) * 2) =
        *reinterpret_cast<const ulonglong4*>(Btile + (long)arow * ldb + k0 + acol);
    __syncthreads();
    mm_stage<0>(smem, wr, wc, lane, acc);
    __syncthreads();
  }
#endif

  // C/D layout: VGPR r holds row (r + 8*(lane>>4)), col = lane&15
  const int rbase = row0 + wr * 32 + 8 * (lane >> 4);
  const int cbase = col0 + wc * 64 + (lane & 15);
#pragma unroll
  for (int s = 0; s < 2; ++s) {
#pragma unroll
    for (int t = 0; t < 4; ++t) {
      const int ccol = cbase + t * 16;
      const float badd = bias ? bias[ccol] : 0.0f;
#pragma unroll
      for (int r = 0; r < 8; ++r) {
        Cb[(long)(rbase + s * 16 + r) * ldc + ccol] = acc[s][t][r] + badd;
      }
    }
  }
}

// ---------------- per-(b,h) max over N*M (key stabilizer) ----------------
__global__ __launch_bounds__(256) void reduce_max_bh(const float* __restrict__ proj,
                                                     float* __restrict__ stab) {
  __shared__ float red[256];
  const float* p = proj + (long)blockIdx.x * N_ * M_;
  float m = -1e30f;
  for (int i = threadIdx.x; i < N_ * M_; i += 256) m = fmaxf(m, p[i]);
  red[threadIdx.x] = m;
  __syncthreads();
  for (int s = 128; s > 0; s >>= 1) {
    if (threadIdx.x < s) red[threadIdx.x] = fmaxf(red[threadIdx.x], red[threadIdx.x + s]);
    __syncthreads();
  }
  if (threadIdx.x == 0) stab[blockIdx.x] = red[0];
}

// ---------------- FAVOR+ finalize: phi = ratio*(exp(proj - norm - stab)+eps) ----------------
__global__ __launch_bounds__(256) void phi_finalize(float* __restrict__ proj,
                                                    const float* __restrict__ yf,
                                                    const float* __restrict__ stab_k,
                                                    int is_query) {
  long token = (long)blockIdx.x * blockDim.x + threadIdx.x;  // 0 .. B*H*N-1
  int bh = (int)(token / N_);
  int n  = (int)(token % N_);
  int b = bh / H_, h = bh % H_;
  const float* y = yf + ((long)(b * N_ + n)) * HD_ + h * D_;
  float s2 = 0.0f;
#pragma unroll 8
  for (int d = 0; d < D_; ++d) {
    float t = y[d] * SCALE_;
    s2 += t * t;
  }
  float norm = 0.5f * s2;
  float* p = proj + token * (long)M_;
  float stab;
  if (is_query) {
    stab = -1e30f;
    for (int m = 0; m < M_; ++m) stab = fmaxf(stab, p[m]);
  } else {
    stab = stab_k[bh];
  }
  for (int m = 0; m < M_; ++m)
    p[m] = RATIO_ * (__expf(p[m] - norm - stab) + EPS_);
}

// ---------------- causal linear-attention prefix scan ----------------
__global__ __launch_bounds__(256) void causal_scan(const float* __restrict__ phi_q,
                                                   const float* __restrict__ phi_k,
                                                   const float* __restrict__ vproj,
                                                   float* __restrict__ out) {
  const int bh = blockIdx.x;
  const int b = bh / H_, h = bh % H_;
  const float* pq = phi_q + (long)bh * N_ * M_;
  const float* pk = phi_k + (long)bh * N_ * M_;
  const int tid = threadIdx.x;
  const int g = tid >> 6;     // m-group 0..3
  const int d = tid & 63;

  __shared__ float s_phik[M_], s_phiq[M_], s_v[D_], s_red[4][D_], s_z[M_], s_den[4];

  float S[32];
#pragma unroll
  for (int j = 0; j < 32; ++j) S[j] = 0.0f;
  if (tid < M_) s_z[tid] = 0.0f;
  __syncthreads();

  for (int n = 0; n < N_; ++n) {
    if (tid < M_) {
      float fk = pk[(long)n * M_ + tid];
      s_phik[tid] = fk;
      s_phiq[tid] = pq[(long)n * M_ + tid];
      s_z[tid] += fk;                         // inclusive k-cumsum
    } else if (tid < M_ + D_) {
      s_v[tid - M_] = vproj[((long)(b * N_ + n)) * HD_ + h * D_ + (tid - M_)];
    }
    __syncthreads();

    const float vd = s_v[d];
    const int mbase = g * 32;
    float num = 0.0f;
#pragma unroll
    for (int j = 0; j < 32; ++j) {
      float s = S[j] + s_phik[mbase + j] * vd;   // inclusive kv-cumsum
      S[j] = s;
      num += s_phiq[mbase + j] * s;
    }
    s_red[g][d] = num;
    if (d == 0) {
      float den = 0.0f;
#pragma unroll
      for (int j = 0; j < 32; ++j) den += s_phiq[mbase + j] * s_z[mbase + j];
      s_den[g] = den;
    }
    __syncthreads();

    if (tid < D_) {
      float acc = s_red[0][tid] + s_red[1][tid] + s_red[2][tid] + s_red[3][tid];
      float den = s_den[0] + s_den[1] + s_den[2] + s_den[3] + EPS_;
      out[((long)(b * N_ + n)) * HD_ + h * D_ + tid] = acc / den;
    }
    __syncthreads();
  }
}

// ---------------- host-side launch ----------------
static inline size_t align256(size_t x) { return (x + 255) & ~(size_t)255; }

extern "C" void kernel_launch(void* const* d_in, const int* in_sizes, int n_in,
                              void* d_out, int out_size, void* d_ws, size_t ws_size,
                              hipStream_t stream) {
  const float* x     = (const float*)d_in[0];
  const float* Wq    = (const float*)d_in[1];
  const float* Wk    = (const float*)d_in[2];
  const float* Wv    = (const float*)d_in[3];
  const float* Wo    = (const float*)d_in[4];
  const float* bo    = (const float*)d_in[5];
  const float* omega = (const float*)d_in[6];
  float* out = (float*)d_out;

  // workspace carve-out
  char* ws = (char*)d_ws;
  size_t off = 0;
  auto take = [&](size_t bytes) { char* p = ws + off; off += align256(bytes); return p; };

  __bf16* xb  = (__bf16*)take((size_t)BN_ * DIM_ * 2);
  __bf16* wqb = (__bf16*)take((size_t)HD_ * DIM_ * 2);
  __bf16* wkb = (__bf16*)take((size_t)HD_ * DIM_ * 2);
  __bf16* wvb = (__bf16*)take((size_t)HD_ * DIM_ * 2);
  __bf16* wob = (__bf16*)take((size_t)DIM_ * HD_ * 2);
  __bf16* omb = (__bf16*)take((size_t)M_ * D_ * 2);
  float*  qf  = (float*)take((size_t)BN_ * HD_ * 4);
  float*  kf  = (float*)take((size_t)BN_ * HD_ * 4);
  float*  vf  = (float*)take((size_t)BN_ * HD_ * 4);
  __bf16* qb  = (__bf16*)take((size_t)BN_ * HD_ * 2);
  __bf16* kb  = (__bf16*)take((size_t)BN_ * HD_ * 2);
  float*  phq = (float*)take((size_t)B_ * H_ * N_ * M_ * 4);
  float*  phk = (float*)take((size_t)B_ * H_ * N_ * M_ * 4);
  float*  stk = (float*)take((size_t)B_ * H_ * 4);
  float*  att = (float*)take((size_t)BN_ * HD_ * 4);
  __bf16* atb = (__bf16*)take((size_t)BN_ * HD_ * 2);

  // 1) precision conversion of activations/weights to bf16
  {
    int n;
    n = BN_ * DIM_; cvt_f32_bf16<<<(n + 255) / 256, 256, 0, stream>>>(x, xb, n, 1.0f);
    n = HD_ * DIM_; cvt_f32_bf16<<<(n + 255) / 256, 256, 0, stream>>>(Wq, wqb, n, 1.0f);
    cvt_f32_bf16<<<(n + 255) / 256, 256, 0, stream>>>(Wk, wkb, n, 1.0f);
    cvt_f32_bf16<<<(n + 255) / 256, 256, 0, stream>>>(Wv, wvb, n, 1.0f);
    cvt_f32_bf16<<<(n + 255) / 256, 256, 0, stream>>>(Wo, wob, n, 1.0f);
    n = M_ * D_;    cvt_f32_bf16<<<(n + 255) / 256, 256, 0, stream>>>(omega, omb, n, 1.0f);
  }

  // 2) Q/K/V projections: [2048,1024] x [1024,1024]^T  (WMMA bf16, TDM-staged)
  {
    dim3 grid(HD_ / TN_, BN_ / TM_, 1);
    gemm_bf16_wmma<<<grid, 256, SMEM_BYTES_, stream>>>(xb, DIM_, wqb, DIM_, qf, HD_, DIM_,
                                                       1, 0, 0, 0, nullptr);
    gemm_bf16_wmma<<<grid, 256, SMEM_BYTES_, stream>>>(xb, DIM_, wkb, DIM_, kf, HD_, DIM_,
                                                       1, 0, 0, 0, nullptr);
    gemm_bf16_wmma<<<grid, 256, SMEM_BYTES_, stream>>>(xb, DIM_, wvb, DIM_, vf, HD_, DIM_,
                                                       1, 0, 0, 0, nullptr);
  }

  // 3) scale + convert q,k for the feature projection
  {
    int n = BN_ * HD_;
    cvt_f32_bf16<<<(n + 255) / 256, 256, 0, stream>>>(qf, qb, n, SCALE_);
    cvt_f32_bf16<<<(n + 255) / 256, 256, 0, stream>>>(kf, kb, n, SCALE_);
  }

  // 4) feature projections proj = (y*scale) @ omega^T per (b,h): batched over z
  {
    dim3 grid(M_ / TN_, N_ / TM_, B_ * H_);
    gemm_bf16_wmma<<<grid, 256, SMEM_BYTES_, stream>>>(qb, HD_, omb, D_, phq, M_, D_,
                                                       H_, (long)N_ * HD_, (long)D_,
                                                       (long)N_ * M_, nullptr);
    gemm_bf16_wmma<<<grid, 256, SMEM_BYTES_, stream>>>(kb, HD_, omb, D_, phk, M_, D_,
                                                       H_, (long)N_ * HD_, (long)D_,
                                                       (long)N_ * M_, nullptr);
  }

  // 5) stabilizers + exp finalize
  reduce_max_bh<<<B_ * H_, 256, 0, stream>>>(phk, stk);
  {
    int tokens = B_ * H_ * N_;
    phi_finalize<<<tokens / 256, 256, 0, stream>>>(phq, qf, nullptr, 1);
    phi_finalize<<<tokens / 256, 256, 0, stream>>>(phk, kf, stk, 0);
  }

  // 6) causal linear-attention scan (register-resident 128x64 state per head)
  causal_scan<<<B_ * H_, 256, 0, stream>>>(phq, phk, vf, att);

  // 7) output projection with fused bias: out = attn @ Wo^T + bo  (WMMA bf16)
  {
    int n = BN_ * HD_;
    cvt_f32_bf16<<<(n + 255) / 256, 256, 0, stream>>>(att, atb, n, 1.0f);
    dim3 grid(DIM_ / TN_, BN_ / TM_, 1);
    gemm_bf16_wmma<<<grid, 256, SMEM_BYTES_, stream>>>(atb, HD_, wob, HD_, out, DIM_, HD_,
                                                       1, 0, 0, 0, bo);
  }
}
